// EquivariantPCR_85074712199223
// MI455X (gfx1250) — matrix operand order repaired and checked by reading, not compiled, not run on hardware
//
#include <hip/hip_runtime.h>
#include <hip/hip_fp16.h>
#include <math.h>

// ---------------------------------------------------------------------------
// EquivariantPCR for MI455X (gfx1250, wave32, WMMA f16 16x16x32 + TDM preload)
//
// Sizes
#define NN   16384
#define CC   32
#define EE   131072
#define NBNB 10
// per-edge output features: k_s(12) | v_s(14) | v_amp(6) = 32 "wi"
// GEMM: hidden32 (K=32) x W2cat (32 x 1024), columns permuted col = wi*32 + c
// ---------------------------------------------------------------------------

typedef _Float16 v16h __attribute__((ext_vector_type(16)));
typedef _Float16 v8h  __attribute__((ext_vector_type(8)));
typedef float    v8f  __attribute__((ext_vector_type(8)));
typedef unsigned int u32x4 __attribute__((ext_vector_type(4)));
typedef int          i32x4 __attribute__((ext_vector_type(4)));
typedef int          i32x8 __attribute__((ext_vector_type(8)));

#if defined(__has_builtin)
#  if __has_builtin(__builtin_amdgcn_tensor_load_to_lds) && \
      __has_builtin(__builtin_amdgcn_s_wait_tensorcnt)
#    define HAVE_TDM 1
#  endif
#endif
#ifndef HAVE_TDM
#  define HAVE_TDM 0
#endif

__device__ __forceinline__ float su_step(float t) {
    return t > 0.f ? __expf(-1.f / t) : 0.f;
}

// Compiler-only scheduling fence for intra-wave LDS producer->consumer.
// Per CDNA5 ISA, same-wave LDS ops execute and retire IN ORDER, and the
// compiler's own DScnt wait before using any later LDS load result
// transitively covers our earlier stores. So no hardware wait is needed,
// only a fence to stop the backend reordering may-alias DS ops.
__device__ __forceinline__ void wave_fence() {
    __builtin_amdgcn_wave_barrier();
}

// ---------------------------------------------------------------------------
// Kernel 0: zero output accumulators and softmax denominators
// ---------------------------------------------------------------------------
__global__ void k_zero(float* __restrict__ out, float* __restrict__ z) {
    int i = blockIdx.x * blockDim.x + threadIdx.x;
    if (i < 2 * NN * 32) out[i] = 0.f;
    if (i < 2 * NN)      z[i]   = 0.f;
}

// ---------------------------------------------------------------------------
// Kernel 1: pack weights.
//  W1cat[b][K] : K<16 -> Wk1[b][K], else Wv1[b][K-16]       (f32, 10x32)
//  Bfrag[t][lane][i] : WMMA-B fragment order, f16, 64 tiles of 16 cols.
//    B element (K,n) for tile t: K = (lane>>4)*16 + i, col' = t*16 + (lane&15)
//    col' = wi*32 + c ;  wi<12: k-scalar  (Wk2[K][c*12+wi],      K<16)
//                        wi<26: v-scalar  (Wv2[K-16][c*14+wi-12], K>=16)
//                        else : v-vector  (Wv2[K-16][448+c*6+wi-26], K>=16)
//    scale 0.25/sqrt(32) folded (1/sqrt(HID) * 1/sqrt(C)).
// ---------------------------------------------------------------------------
__global__ void k_pack(const float* __restrict__ Wk1, const float* __restrict__ Wk2,
                       const float* __restrict__ Wv1, const float* __restrict__ Wv2,
                       _Float16* __restrict__ Bfrag, float* __restrict__ W1cat) {
    int idx = blockIdx.x * blockDim.x + threadIdx.x;
    const float SCL = 0.25f / sqrtf(32.f);
    if (idx < 64 * 512) {
        int t = idx >> 9, rem = idx & 511;
        int l = rem >> 4, i = rem & 15;
        int K = (l >> 4) * 16 + i;
        int colp = t * 16 + (l & 15);
        int wi = colp >> 5, c = colp & 31;
        float v = 0.f;
        if (wi < 12)       { if (K < 16)  v = Wk2[K * 480 + c * 12 + wi]; }
        else if (wi < 26)  { if (K >= 16) v = Wv2[(K - 16) * 640 + c * 14 + (wi - 12)]; }
        else               { if (K >= 16) v = Wv2[(K - 16) * 640 + 448 + c * 6 + (wi - 26)]; }
        Bfrag[idx] = (_Float16)(v * SCL);
    }
    if (idx < 320) {
        int b = idx >> 5, K = idx & 31;
        W1cat[idx] = (K < 16) ? Wk1[b * 16 + K] : Wv1[b * 16 + (K - 16)];
    }
}

// ---------------------------------------------------------------------------
// Kernel 2: per-node qd = (f @ Wq) @ Wdot * 1/sqrt(C) * 1/sqrt(ATTN*KS)
// ---------------------------------------------------------------------------
__global__ void k_qd(const float* __restrict__ x, const float* __restrict__ y,
                     const float* __restrict__ Wq, const float* __restrict__ Wdot,
                     float* __restrict__ qd) {
    int idx = blockIdx.x * blockDim.x + threadIdx.x;
    if (idx >= 2 * NN) return;
    int graph = idx >= NN;
    const float* xf = graph ? y : x;
    int n = idx - graph * NN;
    const float* f = xf + (size_t)n * 35 + 3;
    float q[23];
    for (int a = 0; a < 23; ++a) q[a] = 0.f;
    for (int c = 0; c < 32; ++c) {
        float fc = f[c];
        for (int a = 0; a < 23; ++a) q[a] = fmaf(fc, Wq[c * 23 + a], q[a]);
    }
    const float QS = 1.f / (sqrtf(32.f) * sqrtf(23.f * 12.f));
    float* o = qd + (size_t)idx * 12;
    for (int wi = 0; wi < 12; ++wi) {
        float s = 0.f;
        for (int a = 0; a < 23; ++a) s = fmaf(q[a], Wdot[a * 12 + wi], s);
        o[wi] = s * QS;
    }
}

// ---------------------------------------------------------------------------
// Kernel 3: heavy edge pass. One wave processes 16 edges (one WMMA M-tile).
// Dynamic LDS layout (bytes): Bs 65536 | W1s 1280 | hid 4x1024(f16) |
//                             wst 4x2048(f32) | outs 4x2048(f32)  = 87296
// ---------------------------------------------------------------------------
__global__ __launch_bounds__(128) void k_edge(
    const float* __restrict__ xin, const float* __restrict__ yin,
    const int* __restrict__ sxA, const int* __restrict__ dxA,
    const int* __restrict__ syA, const int* __restrict__ dyA,
    const _Float16* __restrict__ BfragG, const float* __restrict__ W1G,
    const float* __restrict__ qd, float* __restrict__ zacc,
    float* __restrict__ expw, float* __restrict__ vout) {
    extern __shared__ char smem[];
    _Float16* Bs   = (_Float16*)smem;               // 65536 B (LDS offset 0)
    float*    W1s  = (float*)(smem + 65536);        //  1280 B
    _Float16* hidA = (_Float16*)(smem + 66816);     //  4096 B
    float*    wstA = (float*)(smem + 70912);        //  8192 B
    float*    outA = (float*)(smem + 79104);        //  8192 B

    const int tid = threadIdx.x;
    const int wv  = tid >> 5;

#if HAVE_TDM
    // TDM preload: one DMA of the 64KB B-fragment block into LDS offset 0
    // (dynamic LDS starts at offset 0; no static LDS in this kernel).
    // D# describes a 1D tensor: 8192 x 8B elements, tile = whole row.
    if (wv == 0) {
        unsigned long long ga = (unsigned long long)(uintptr_t)BfragG;
        u32x4 g0;
        g0[0] = 1u;                                   // count=1, user descriptor
        g0[1] = 0u;                                   // lds_addr = 0
        g0[2] = (unsigned int)(ga & 0xffffffffu);     // global_addr[31:0]
        g0[3] = (unsigned int)((ga >> 32) & 0x1ffffffu) | (2u << 30); // +type=2
        i32x8 g1;
        g1[0] = (3 << 16);                 // wg_mask=0, data_size=8B
        g1[1] = (int)(8192u << 16);        // tensor_dim0[15:0] in bits [63:48]
        g1[2] = (1 << 16);                 // tensor_dim0 hi=0, tensor_dim1=1
        g1[3] = (int)(8192u << 16);        // tensor_dim1 hi=0, tile_dim0=8192
        g1[4] = 1;                         // tile_dim1=1, tile_dim2=0
        g1[5] = 8192;                      // tensor_dim0_stride lo
        g1[6] = 0;
        g1[7] = 0;
        i32x4 zz4 = {0, 0, 0, 0};
#  if __clang_major__ >= 23
        i32x8 zz8 = {0, 0, 0, 0, 0, 0, 0, 0};
        __builtin_amdgcn_tensor_load_to_lds(g0, g1, zz4, zz4, zz8, 0);
#  else
        __builtin_amdgcn_tensor_load_to_lds(g0, g1, zz4, zz4, 0);
#  endif
        __builtin_amdgcn_s_wait_tensorcnt(0);
    }
#else
    {   // fallback: block-cooperative preload of B fragments
        const int4* s4 = (const int4*)BfragG;
        int4* d4 = (int4*)Bs;
        for (int i = tid; i < 65536 / 16; i += 128) d4[i] = s4[i];
    }
#endif
    for (int i = tid; i < 320; i += 128) W1s[i] = W1G[i];
    __syncthreads();

    const int lane = tid & 31;
    const int kh   = lane >> 4;   // half: A K-block select / hidden half / c-half
    const int le   = lane & 15;   // edge-in-group == A row == B column-in-tile
    _Float16* hid  = hidA + wv * 512;
    float*    wst  = wstA + wv * 512;
    float*    outs = outA + wv * 512;

    const int GPG = EE / 16;
    const int nwaves = gridDim.x * 4;
    for (int g = blockIdx.x * 4 + wv; g < 2 * GPG; g += nwaves) {
        const int graph = (g >= GPG) ? 1 : 0;
        const float* xf  = graph ? yin : xin;
        const int* srcA  = graph ? syA : sxA;
        const int* dstA  = graph ? dyA : dxA;
        const int e   = (g - graph * GPG) * 16 + le;
        const int src = srcA[e];
        const int dst = dstA[e];
        const float* rs = xf + (size_t)src * 35;
        const float* rd = xf + (size_t)dst * 35;
        const float ex = rs[0] - rd[0], ey = rs[1] - rd[1], ez = rs[2] - rd[2];
        const float elen = sqrtf(ex * ex + ey * ey + ez * ez);
        const float rinv = 1.f / fmaxf(elen, 1e-9f);
        const float S3 = 1.7320508075688772f;
        const float sh1v0 = S3 * ey * rinv, sh1v1 = S3 * ez * rinv, sh1v2 = S3 * ex * rinv;
        const float cutoff = su_step(10.f * (1.f - elen * 0.5f));
        // soft one-hot basis (sqrt(NB) factors cancel between emb and /sqrt(NB))
        float emb[NBNB];
        {
            const float stepv = 2.f / 11.f, istep = 11.f / 2.f;
            const float pref = 1.14136f * 7.3890561f;   // 1.14136 * e^2
            #pragma unroll
            for (int b = 0; b < NBNB; ++b) {
                float d = (elen - stepv * (b + 1)) * istep;
                emb[b] = pref * su_step(d + 1.f) * su_step(1.f - d);
            }
        }
        // fs slice: lane holds f[src][kh*16 .. kh*16+15]
        float fs[16];
        {
            const float* fsp = rs + 3 + kh * 16;
            #pragma unroll
            for (int j = 0; j < 16; ++j) fs[j] = fsp[j];
        }
        // hidden half: lane computes K = kh*16 + h, h=0..15 ; silu activation
        {
            float hacc[16];
            #pragma unroll
            for (int h = 0; h < 16; ++h) hacc[h] = 0.f;
            #pragma unroll
            for (int b = 0; b < NBNB; ++b) {
                const float wb = emb[b];
                const float* wrow = W1s + b * 32 + kh * 16;
                #pragma unroll
                for (int h = 0; h < 16; ++h) hacc[h] = fmaf(wb, wrow[h], hacc[h]);
            }
            _Float16* hrow = hid + le * 32 + kh * 16;
            #pragma unroll
            for (int h = 0; h < 16; ++h) {
                float xh = hacc[h];
                hrow[h] = (_Float16)(xh / (1.f + __expf(-xh)));
            }
        }
        wave_fence();
        // A fragment (contiguous K order, matches Bfrag): lane row M=le,
        // elements 0..15 <- K = 16*kh..16*kh+15
        v16h afrag;
        {
            v8h lo = *(const v8h*)(hid + le * 32 + 16 * kh);
            v8h hi = *(const v8h*)(hid + le * 32 + 16 * kh + 8);
            #pragma unroll
            for (int i = 0; i < 8; ++i) { afrag[i] = lo[i]; afrag[8 + i] = hi[i]; }
        }
        // 32 output features; each is a tile-pair (c=0..15, c=16..31)
        for (int wi = 0; wi < 32; ++wi) {
            v16h b0, b1;
            {
                const v8h* p0 = (const v8h*)(Bs + (size_t)(2 * wi) * 512 + lane * 16);
                const v8h* p1 = (const v8h*)(Bs + (size_t)(2 * wi + 1) * 512 + lane * 16);
                v8h a0 = p0[0], a1 = p0[1], c0 = p1[0], c1 = p1[1];
                #pragma unroll
                for (int i = 0; i < 8; ++i) {
                    b0[i] = a0[i]; b0[8 + i] = a1[i];
                    b1[i] = c0[i]; b1[8 + i] = c1[i];
                }
            }
            v8f cz = {};
            v8f acc0 = __builtin_amdgcn_wmma_f32_16x16x32_f16(
                false, afrag, false, b0, (short)0, cz, false, false);
            v8f acc1 = __builtin_amdgcn_wmma_f32_16x16x32_f16(
                false, afrag, false, b1, (short)0, cz, false, false);
            // D layout: lane l, reg r -> row M = r + 8*(l>>4), col N = l&15
            #pragma unroll
            for (int r = 0; r < 8; ++r) {
                wst[(r + 8 * kh) * 32 + le]      = acc0[r];
                wst[(r + 8 * kh) * 32 + 16 + le] = acc1[r];
            }
            wave_fence();
            // contract with fs over c: lane handles edge le, c in [kh*16, kh*16+16)
            const float4* pw4 = (const float4*)(wst + le * 32 + kh * 16);
            float4 wA = pw4[0], wB = pw4[1], wC = pw4[2], wD = pw4[3];
            float s = 0.f;
            s = fmaf(wA.x, fs[0],  s); s = fmaf(wA.y, fs[1],  s);
            s = fmaf(wA.z, fs[2],  s); s = fmaf(wA.w, fs[3],  s);
            s = fmaf(wB.x, fs[4],  s); s = fmaf(wB.y, fs[5],  s);
            s = fmaf(wB.z, fs[6],  s); s = fmaf(wB.w, fs[7],  s);
            s = fmaf(wC.x, fs[8],  s); s = fmaf(wC.y, fs[9],  s);
            s = fmaf(wC.z, fs[10], s); s = fmaf(wC.w, fs[11], s);
            s = fmaf(wD.x, fs[12], s); s = fmaf(wD.y, fs[13], s);
            s = fmaf(wD.z, fs[14], s); s = fmaf(wD.w, fs[15], s);
            s += __shfl_xor(s, 16, 32);
            if (kh == 0) outs[le * 32 + wi] = s;
            wave_fence();
        }
        wave_fence();
        // logits / softmax numerator (lanes 0..15, one per edge)
        if (kh == 0) {
            const float* qrow = qd + (size_t)(graph * NN + dst) * 12;
            float lg = 0.f;
            #pragma unroll
            for (int w2 = 0; w2 < 12; ++w2) lg = fmaf(qrow[w2], outs[le * 32 + w2], lg);
            float expv = cutoff * __expf(lg);
            expw[(size_t)graph * EE + e] = expv;
            atomicAdd(zacc + (size_t)graph * NN + dst, expv);
        }
        // edge value vector: v_s(14) then amp[j]*sh1[a] laid out j*3+a
        {
            float val[16];
            #pragma unroll
            for (int qq = 0; qq < 16; ++qq) {
                int col = kh * 16 + qq;
                if (col < 14) val[qq] = outs[le * 32 + 12 + col];
                else {
                    int t2 = col - 14;
                    int j = t2 / 3, a = t2 - 3 * j;
                    float sh1a = (a == 0) ? sh1v0 : (a == 1) ? sh1v1 : sh1v2;
                    val[qq] = outs[le * 32 + 26 + j] * sh1a;
                }
            }
            // vout row is 128B aligned; lane writes 64B contiguous
            float4* vo4 = (float4*)(vout + ((size_t)graph * EE + e) * 32 + kh * 16);
            vo4[0] = make_float4(val[0],  val[1],  val[2],  val[3]);
            vo4[1] = make_float4(val[4],  val[5],  val[6],  val[7]);
            vo4[2] = make_float4(val[8],  val[9],  val[10], val[11]);
            vo4[3] = make_float4(val[12], val[13], val[14], val[15]);
        }
        wave_fence();
    }
}

// ---------------------------------------------------------------------------
// Kernel 4: softmax normalize + scatter-add into output
// ---------------------------------------------------------------------------
__global__ void k_scatter(const int* __restrict__ dxA, const int* __restrict__ dyA,
                          const float* __restrict__ expw, const float* __restrict__ zacc,
                          const float* __restrict__ vout, float* __restrict__ out) {
    int tid = blockIdx.x * blockDim.x + threadIdx.x;
    if (tid >= 2 * EE * 4) return;
    int eg = tid >> 2, part = tid & 3;
    int graph = eg >= EE;
    int e = eg - graph * EE;
    int dst = (graph ? dyA : dxA)[e];
    float ew = expw[eg];
    float zz = zacc[graph * NN + dst];
    zz = (zz == 0.f) ? 1.f : zz;
    float coef = sqrtf(ew / zz + 1e-20f);
    const float* vv = vout + (size_t)eg * 32 + part * 8;
    float* oo = out + ((size_t)(graph * NN + dst)) * 32 + part * 8;
    #pragma unroll
    for (int j = 0; j < 8; ++j) atomicAdd(oo + j, coef * vv[j]);
}

// ---------------------------------------------------------------------------
extern "C" void kernel_launch(void* const* d_in, const int* in_sizes, int n_in,
                              void* d_out, int out_size, void* d_ws, size_t ws_size,
                              hipStream_t stream) {
    (void)in_sizes; (void)n_in; (void)out_size; (void)ws_size;
    const float* x    = (const float*)d_in[0];
    const float* y    = (const float*)d_in[1];
    const int*   sx   = (const int*)d_in[2];
    const int*   dx   = (const int*)d_in[3];
    const int*   sy   = (const int*)d_in[4];
    const int*   dy   = (const int*)d_in[5];
    const float* Wq   = (const float*)d_in[6];
    const float* Wk1  = (const float*)d_in[7];
    const float* Wk2  = (const float*)d_in[8];
    const float* Wv1  = (const float*)d_in[9];
    const float* Wv2  = (const float*)d_in[10];
    const float* Wdot = (const float*)d_in[11];
    float* out = (float*)d_out;
    char*  ws  = (char*)d_ws;

    // workspace layout (~36.4 MB)
    _Float16* Bfrag = (_Float16*)(ws);               //    65536 B
    float* W1cat = (float*)(ws + 65536);             //     1280 B
    float* qd    = (float*)(ws + 66816);             //  1572864 B
    float* z     = (float*)(ws + 1639680);           //   131072 B
    float* expw  = (float*)(ws + 1770752);           //  1048576 B
    float* vout  = (float*)(ws + 2819328);           // 33554432 B

    (void)hipFuncSetAttribute((const void*)k_edge,
                              hipFuncAttributeMaxDynamicSharedMemorySize, 87296);

    k_zero<<<(2 * NN * 32) / 256, 256, 0, stream>>>(out, z);
    k_pack<<<(64 * 512) / 256, 256, 0, stream>>>(Wk1, Wk2, Wv1, Wv2, Bfrag, W1cat);
    k_qd<<<(2 * NN) / 256, 256, 0, stream>>>(x, y, Wq, Wdot, qd);
    k_edge<<<1024, 128, 87296, stream>>>(x, y, sx, dx, sy, dy,
                                         Bfrag, W1cat, qd, z, expw, vout);
    k_scatter<<<(2 * EE * 4) / 256, 256, 0, stream>>>(dx, dy, expw, z, vout, out);
}
